// GPRNet_7576322310700
// MI455X (gfx1250) — matrix operand order repaired and checked by reading, not compiled, not run on hardware
//
#include <hip/hip_runtime.h>
#include <hip/hip_bf16.h>

typedef __attribute__((ext_vector_type(16))) _Float16 v16h;
typedef __attribute__((ext_vector_type(8)))  float    v8f;

#define FDIM 64
#define HDIM 32

// ---------------------------------------------------------------- utilities
__device__ __forceinline__ void atomAddF(float* p, float v) {
    // relaxed, device(agent) scope -> global_atomic_add_f32 (L2 RMW)
    __hip_atomic_fetch_add(p, v, __ATOMIC_RELAXED, __HIP_MEMORY_SCOPE_AGENT);
}

// ---------------------------------------------------------------- degree / norm
__global__ __launch_bounds__(256) void k_fill1(float* __restrict__ deg, int n) {
    int i = blockIdx.x * blockDim.x + threadIdx.x;
    if (i < n) deg[i] = 1.0f;                      // self-loop contributes 1
}

__global__ __launch_bounds__(256) void k_degree(const int* __restrict__ col,
                                                float* __restrict__ deg, int E) {
    int e = blockIdx.x * blockDim.x + threadIdx.x;
    if (e < E) atomAddF(deg + col[e], 1.0f);
}

__global__ __launch_bounds__(256) void k_dinv(const float* __restrict__ deg,
                                              float* __restrict__ dinv,
                                              float* __restrict__ selfn, int n) {
    int i = blockIdx.x * blockDim.x + threadIdx.x;
    if (i >= n) return;
    float d = deg[i];                              // always >= 1
    float r = rsqrtf(d);
    dinv[i]  = r;
    selfn[i] = r * r;                              // dinv[i]*dinv[i] (self-loop weight)
}

__global__ __launch_bounds__(256) void k_norm(const int* __restrict__ row,
                                              const int* __restrict__ col,
                                              const float* __restrict__ dinv,
                                              float* __restrict__ norm, int E) {
    int e = blockIdx.x * blockDim.x + threadIdx.x;
    if (e < E) norm[e] = dinv[row[e]] * dinv[col[e]];
}

// ---------------------------------------------------------------- fused MLP via WMMA
// h  = relu(relu(x*W1+b1) @ W2 + b2)   [N,64]
// hidden = temp[0] * h
// One wave handles a 16-node tile; K=32 fits a single v_wmma_f32_16x16x32_f16.
__global__ __launch_bounds__(64) void k_mlp_wmma(const float* __restrict__ x,
                                                 const float* __restrict__ W1,
                                                 const float* __restrict__ b1,
                                                 const float* __restrict__ W2,
                                                 const float* __restrict__ b2,
                                                 const float* __restrict__ temp,
                                                 float* __restrict__ h,
                                                 float* __restrict__ hidden,
                                                 int ntiles) {
    const int wave = threadIdx.x >> 5;                 // 2 waves / block
    const int tile = blockIdx.x * 2 + wave;
    if (tile >= ntiles) return;                        // wave-uniform: EXEC stays all-1s
    const int lane = threadIdx.x & 31;
    const int m    = lane & 15;                        // row within tile (A) / col (B,C,D)
    const int hi   = lane >> 4;                        // lane-half select
    const int node = tile * 16 + m;

    // A fragment: A[m][k] = relu(x[node]*W1[k] + b1[k]), 16x32 f16
    // half-index i: i<8 -> K = 8*hi + i ; i>=8 -> K = 16 + 8*hi + (i-8)
    const float xv = x[node];
    v16h a;
#pragma unroll
    for (int i = 0; i < 16; ++i) {
        int k = (i < 8) ? (8 * hi + i) : (16 + 8 * hi + (i - 8));
        float v = fmaf(xv, W1[k], b1[k]);
        a[i] = (_Float16)(v > 0.0f ? v : 0.0f);
    }

    const float t0 = temp[0];

    // 4 column tiles of 16 cover the 64 output features
#pragma unroll
    for (int ct = 0; ct < 4; ++ct) {
        const int ncol = ct * 16 + m;
        // B fragment: B[k][n], half-index i: i<8 -> K = 8*hi + i ; i>=8 -> K = 16 + 8*hi + (i-8), N = ncol
        v16h b;
#pragma unroll
        for (int i = 0; i < 16; ++i) {
            int k = (i < 8) ? (8 * hi + i) : (16 + 8 * hi + (i - 8));
            b[i] = (_Float16)W2[k * FDIM + ncol];
        }
        // C preloaded with bias b2[ncol] (same N for all 8 row-slots of this lane)
        v8f c;
        const float bias = b2[ncol];
#pragma unroll
        for (int v = 0; v < 8; ++v) c[v] = bias;

        c = __builtin_amdgcn_wmma_f32_16x16x32_f16(
                /*neg_a=*/false, a, /*neg_b=*/false, b,
                /*c_mod=*/(short)0, c, /*reuse_a=*/false, /*reuse_b=*/false);

        // D layout: VGPR v -> M = v + 8*hi, N = ncol. Fuse ReLU + temp[0] scaling.
#pragma unroll
        for (int v = 0; v < 8; ++v) {
            const int rowN = tile * 16 + v + 8 * hi;
            float hv = c[v] > 0.0f ? c[v] : 0.0f;
            const size_t o = (size_t)rowN * FDIM + ncol;
            h[o]      = hv;
            hidden[o] = t0 * hv;
        }
    }
}

// scalar tail for N % 16 leftover nodes (not hit for N=100000, kept for generality)
__global__ __launch_bounds__(64) void k_mlp_tail(const float* __restrict__ x,
                                                 const float* __restrict__ W1,
                                                 const float* __restrict__ b1,
                                                 const float* __restrict__ W2,
                                                 const float* __restrict__ b2,
                                                 const float* __restrict__ temp,
                                                 float* __restrict__ h,
                                                 float* __restrict__ hidden,
                                                 int start, int n) {
    int node = start + blockIdx.x * blockDim.x + threadIdx.x;
    if (node >= n) return;
    float h1[HDIM];
    const float xv = x[node];
#pragma unroll
    for (int k = 0; k < HDIM; ++k) {
        float v = fmaf(xv, W1[k], b1[k]);
        h1[k] = v > 0.0f ? v : 0.0f;
    }
    const float t0 = temp[0];
    for (int c = 0; c < FDIM; ++c) {
        float acc = b2[c];
#pragma unroll
        for (int k = 0; k < HDIM; ++k) acc = fmaf(h1[k], W2[k * FDIM + c], acc);
        acc = acc > 0.0f ? acc : 0.0f;
        const size_t o = (size_t)node * FDIM + c;
        h[o]      = acc;
        hidden[o] = t0 * acc;
    }
}

// ---------------------------------------------------------------- propagation hop
// x_next = selfn .* x_cur   (self-loop term; also initializes scatter target)
__global__ __launch_bounds__(256) void k_initnext(float* __restrict__ xn,
                                                  const float* __restrict__ xc,
                                                  const float* __restrict__ selfn,
                                                  int n) {
    int t = blockIdx.x * blockDim.x + threadIdx.x;    // one float4 per thread
    if (t >= n * (FDIM / 4)) return;
    const int node = t >> 4;
    const float s = selfn[node];
    float4 v = ((const float4*)xc)[t];
    v.x *= s; v.y *= s; v.z *= s; v.w *= s;
    ((float4*)xn)[t] = v;
}

// x_next[col] += norm[e] * x_cur[row] ; 16 lanes per edge, float4 per lane
__global__ __launch_bounds__(256) void k_scatter(const int* __restrict__ row,
                                                 const int* __restrict__ col,
                                                 const float* __restrict__ norm,
                                                 const float* __restrict__ xc,
                                                 float* __restrict__ xn, int E) {
    long long t = (long long)blockIdx.x * blockDim.x + threadIdx.x;
    long long e = t >> 4;
    if (e >= E) return;
    const int  q = (int)(t & 15);
    const int  r = row[e];
    const int  c = col[e];
    const float w = norm[e];
    const float4 v = ((const float4*)(xc + (size_t)r * FDIM))[q];
    float* dst = xn + (size_t)c * FDIM + q * 4;
    atomAddF(dst + 0, w * v.x);
    atomAddF(dst + 1, w * v.y);
    atomAddF(dst + 2, w * v.z);
    atomAddF(dst + 3, w * v.w);
}

// hidden += temp[k] * x_next
__global__ __launch_bounds__(256) void k_accum(float* __restrict__ hidden,
                                               const float* __restrict__ xn,
                                               const float* __restrict__ temp,
                                               int kidx, int n) {
    int t = blockIdx.x * blockDim.x + threadIdx.x;
    if (t >= n * (FDIM / 4)) return;
    const float g = temp[kidx];
    float4 hv = ((const float4*)hidden)[t];
    float4 xv = ((const float4*)xn)[t];
    hv.x = fmaf(g, xv.x, hv.x);
    hv.y = fmaf(g, xv.y, hv.y);
    hv.z = fmaf(g, xv.z, hv.z);
    hv.w = fmaf(g, xv.w, hv.w);
    ((float4*)hidden)[t] = hv;
}

// ---------------------------------------------------------------- output projection
__global__ __launch_bounds__(256) void k_out(const float* __restrict__ hidden,
                                             const float* __restrict__ Wfc,
                                             const float* __restrict__ bfc,
                                             float* __restrict__ out, int n) {
    int node = blockIdx.x * blockDim.x + threadIdx.x;
    if (node >= n) return;
    const float4* hp = (const float4*)(hidden + (size_t)node * FDIM);
    const float4* wp = (const float4*)Wfc;
    float acc = 0.0f;
#pragma unroll
    for (int i = 0; i < FDIM / 4; ++i) {
        float4 a = hp[i], b = wp[i];
        acc = fmaf(a.x, b.x, acc);
        acc = fmaf(a.y, b.y, acc);
        acc = fmaf(a.z, b.z, acc);
        acc = fmaf(a.w, b.w, acc);
    }
    out[node] = acc + bfc[0];
}

// ---------------------------------------------------------------- launcher
extern "C" void kernel_launch(void* const* d_in, const int* in_sizes, int n_in,
                              void* d_out, int out_size, void* d_ws, size_t ws_size,
                              hipStream_t stream) {
    const float* x     = (const float*)d_in[0];
    const int*   eidx  = (const int*)  d_in[1];
    const float* W1    = (const float*)d_in[2];
    const float* b1    = (const float*)d_in[3];
    const float* W2    = (const float*)d_in[4];
    const float* b2    = (const float*)d_in[5];
    const float* temp  = (const float*)d_in[6];
    const float* Wfc   = (const float*)d_in[7];
    const float* bfc   = (const float*)d_in[8];
    float*       out   = (float*)d_out;

    const int N = in_sizes[0];              // x is [N,1]
    const int E = in_sizes[1] / 2;          // edge_index is [2,E]
    const int K = in_sizes[6] - 1;          // temp is [K+1]

    const int* row = eidx;                  // sources
    const int* col = eidx + E;              // targets (aggregation index)

    // workspace layout (floats)
    const size_t NF = (size_t)N * FDIM;
    float* w      = (float*)d_ws;
    float* hbuf   = w;                      // [N,64]  x_0 / ping
    float* pbuf   = hbuf + NF;              // [N,64]  pong
    float* hidden = pbuf + NF;              // [N,64]
    float* deg    = hidden + NF;            // [N]
    float* dinv   = deg + N;                // [N]
    float* selfn  = dinv + N;               // [N]
    float* norm   = selfn + N;              // [E]

    const int B = 256;
    const int gN  = (N + B - 1) / B;
    const int gE  = (E + B - 1) / B;
    const int gNF4 = (N * (FDIM / 4) + B - 1) / B;
    const long long scatThreads = (long long)E * 16;
    const int gScat = (int)((scatThreads + B - 1) / B);

    // ---- gcn_norm (self-loops folded into deg init / selfn) ----
    k_fill1 <<<gN, B, 0, stream>>>(deg, N);
    k_degree<<<gE, B, 0, stream>>>(col, deg, E);
    k_dinv  <<<gN, B, 0, stream>>>(deg, dinv, selfn, N);
    k_norm  <<<gE, B, 0, stream>>>(row, col, dinv, norm, E);

    // ---- feature MLP (WMMA) + hidden = temp[0]*h ----
    const int ntiles = N / 16;
    if (ntiles > 0)
        k_mlp_wmma<<<(ntiles + 1) / 2, 64, 0, stream>>>(x, W1, b1, W2, b2, temp,
                                                        hbuf, hidden, ntiles);
    const int rem = N - ntiles * 16;
    if (rem > 0)
        k_mlp_tail<<<(rem + 63) / 64, 64, 0, stream>>>(x, W1, b1, W2, b2, temp,
                                                       hbuf, hidden, ntiles * 16, N);

    // ---- K-hop GPR propagation ----
    float* cur = hbuf;
    float* nxt = pbuf;
    for (int k = 1; k <= K; ++k) {
        k_initnext<<<gNF4, B, 0, stream>>>(nxt, cur, selfn, N);
        k_scatter <<<gScat, B, 0, stream>>>(row, col, norm, cur, nxt, E);
        k_accum   <<<gNF4, B, 0, stream>>>(hidden, nxt, temp, k, N);
        float* t = cur; cur = nxt; nxt = t;
    }

    // ---- output projection ----
    k_out<<<gN, B, 0, stream>>>(hidden, Wfc, bfc, out, N);
}
